// GridMultiHeadAttention_2877628088558
// MI455X (gfx1250) — matrix operand order repaired and checked by reading, not compiled, not run on hardware
//
#include <hip/hip_runtime.h>
#include <hip/hip_bf16.h>

// Problem constants (match reference)
#define BB 4
#define SS 2048
#define HH 1024
#define NH 8
#define HD 128
#define MROWS (BB * SS) // 8192 total rows

typedef __attribute__((ext_vector_type(16))) __bf16 bf16x16;
typedef __attribute__((ext_vector_type(8))) float f32x8;

union Frag {
    bf16x16 bf;
    uint4 q[2];
};

union Pack {
    unsigned short h[8];
    uint4 q;
};

__device__ __forceinline__ unsigned short f2bf(float f) {
    unsigned int u = __builtin_bit_cast(unsigned int, f);
    u += 0x7FFFu + ((u >> 16) & 1u); // round-to-nearest-even
    return (unsigned short)(u >> 16);
}

// A fragment (16x32, M x K), row-major source, leading dim `ld` (elements).
// lane l holds row (l&15); lanes 0-15 hold K 0..15, lanes 16-31 hold K 16..31.
__device__ __forceinline__ Frag load_a_frag(const unsigned short* __restrict__ base,
                                            int ld, int row0, int col0, int lane) {
    const int lh = lane & 15, hf = lane >> 4;
    const unsigned short* p = base + (size_t)(row0 + lh) * ld + col0 + hf * 16;
    Frag f;
    f.q[0] = *(const uint4*)(p);
    f.q[1] = *(const uint4*)(p + 8);
    return f;
}

// B fragment (32x16, K x N); column n contiguous at base + n*ld.
// lane l holds column (l&15); lanes 0-15 hold K 0..15, lanes 16-31 hold K 16..31.
__device__ __forceinline__ Frag load_b_frag(const unsigned short* __restrict__ base,
                                            int ld, int col0, int k0, int lane) {
    const int lh = lane & 15, hf = lane >> 4;
    const unsigned short* p = base + (size_t)(col0 + lh) * ld + k0 + hf * 16;
    Frag f;
    f.q[0] = *(const uint4*)(p);
    f.q[1] = *(const uint4*)(p + 8);
    return f;
}

__device__ __forceinline__ f32x8 wmma_bf16(const Frag& a, const Frag& b, f32x8 c) {
    return __builtin_amdgcn_wmma_f32_16x16x32_bf16(false, a.bf, false, b.bf,
                                                   (short)0, c, false, false);
}

// ---------------- Kernel 0: f32 -> bf16 ----------------
__global__ void k_convert_bf16(const float* __restrict__ in,
                               unsigned short* __restrict__ out, int n) {
    int i = blockIdx.x * blockDim.x + threadIdx.x;
    const int stride = gridDim.x * blockDim.x;
    for (; i < n; i += stride) out[i] = f2bf(in[i]);
}

// ---------------- Kernel 1: fused QKV projection ----------------
// Q/K: transposed compute (A = W feature rows, B = x^T token cols) -> packed b128 stores [b,h,s,d]
// V:   normal compute -> packed b128 stores into transposed [b,h,d,s]
__global__ void __launch_bounds__(256) k_qkv(
    const unsigned short* __restrict__ xb,
    const unsigned short* __restrict__ Wqb, const unsigned short* __restrict__ Wkb,
    const unsigned short* __restrict__ Wvb,
    const float* __restrict__ bq, const float* __restrict__ bk, const float* __restrict__ bv,
    unsigned short* __restrict__ Qh, unsigned short* __restrict__ Kh,
    unsigned short* __restrict__ Vt) {
    const int lane = threadIdx.x & 31;
    const int wid = (int)((blockIdx.x * blockDim.x + threadIdx.x) >> 5);
    const int mat = wid >> 13;      // 8192 waves per matrix
    const int rem = wid & 8191;
    const int m0 = (rem >> 4) << 4; // token tile base
    const int n0 = (rem & 15) << 6; // feature tile base (64 wide)
    const int lh = lane & 15, hf = lane >> 4;

    if (mat < 2) {
        const unsigned short* W = (mat == 0) ? Wqb : Wkb;
        const float* bias = (mat == 0) ? bq : bk;
        unsigned short* Out = (mat == 0) ? Qh : Kh;

        f32x8 acc[4] = {};
        Frag bx = load_b_frag(xb, HH, m0, 0, lane);
        Frag aW[4];
#pragma unroll
        for (int jb = 0; jb < 4; ++jb) aW[jb] = load_a_frag(W, HH, n0 + jb * 16, 0, lane);
        for (int kc = 0; kc < HH; kc += 32) {
            const int kn = (kc + 32 < HH) ? (kc + 32) : 0;
            Frag bxn = load_b_frag(xb, HH, m0, kn, lane);
            Frag aWn[4];
#pragma unroll
            for (int jb = 0; jb < 4; ++jb) aWn[jb] = load_a_frag(W, HH, n0 + jb * 16, kn, lane);
#pragma unroll
            for (int jb = 0; jb < 4; ++jb) acc[jb] = wmma_bf16(aW[jb], bx, acc[jb]);
            bx = bxn;
#pragma unroll
            for (int jb = 0; jb < 4; ++jb) aW[jb] = aWn[jb];
        }
        // lane = one token; 8 consecutive features per tile -> packed store
        const int m = m0 + lh;
        const int b_ = m >> 11, s = m & 2047;
#pragma unroll
        for (int jb = 0; jb < 4; ++jb) {
            const int nb = n0 + jb * 16 + hf * 8; // first of 8 consecutive features
            const int head = nb >> 7, d0 = nb & 127;
            Pack pk;
#pragma unroll
            for (int j = 0; j < 8; ++j) pk.h[j] = f2bf(acc[jb][j] + bias[nb + j]);
            *(uint4*)&Out[(((size_t)b_ * NH + head) * SS + s) * HD + d0] = pk.q;
        }
    } else {
        f32x8 acc[4] = {};
        Frag a = load_a_frag(xb, HH, m0, 0, lane);
        Frag bw[4];
#pragma unroll
        for (int jb = 0; jb < 4; ++jb) bw[jb] = load_b_frag(Wvb, HH, n0 + jb * 16, 0, lane);
        for (int kc = 0; kc < HH; kc += 32) {
            const int kn = (kc + 32 < HH) ? (kc + 32) : 0;
            Frag an = load_a_frag(xb, HH, m0, kn, lane);
            Frag bwn[4];
#pragma unroll
            for (int jb = 0; jb < 4; ++jb) bwn[jb] = load_b_frag(Wvb, HH, n0 + jb * 16, kn, lane);
#pragma unroll
            for (int jb = 0; jb < 4; ++jb) acc[jb] = wmma_bf16(a, bw[jb], acc[jb]);
            a = an;
#pragma unroll
            for (int jb = 0; jb < 4; ++jb) bw[jb] = bwn[jb];
        }
        // lane = one feature; 8 consecutive tokens per tile -> packed store into Vt
        const int b_ = m0 >> 11;
        const int s0_ = (m0 & 2047) + hf * 8;
#pragma unroll
        for (int jb = 0; jb < 4; ++jb) {
            const int n = n0 + jb * 16 + lh;
            const int head = n >> 7, d = n & 127;
            const float bn = bv[n];
            Pack pk;
#pragma unroll
            for (int j = 0; j < 8; ++j) pk.h[j] = f2bf(acc[jb][j] + bn);
            *(uint4*)&Vt[(((size_t)b_ * NH + head) * HD + d) * SS + s0_] = pk.q;
        }
    }
}

// ---------------- Kernel 2: flash attention (token = lane orientation) ----------------
// S^T = K * Q^T ; per-lane online softmax ; O^T = V^T * P^T ; packed stores throughout
__global__ void __launch_bounds__(128) k_attn(const unsigned short* __restrict__ Qh,
                                              const unsigned short* __restrict__ Kh,
                                              const unsigned short* __restrict__ Vt,
                                              unsigned short* __restrict__ ctx) {
    __shared__ unsigned short plds[4][16 * 32]; // per-wave 16 tokens x 32 keys bf16
    const int lane = threadIdx.x & 31;
    const int wIn = threadIdx.x >> 5;
    const int wid = blockIdx.x * 4 + wIn;
    const int bh = wid >> 7;  // (batch,head) 0..31
    const int qt = wid & 127; // 16-token q tile
    const int m0 = qt * 16;
    const int lh = lane & 15, hf = lane >> 4;

    const unsigned short* Qp = Qh + (size_t)bh * SS * HD;
    const unsigned short* Kp = Kh + (size_t)bh * SS * HD;
    const unsigned short* Vp = Vt + (size_t)bh * HD * SS;

    // Q tokens as B-operand columns; resident for the whole key loop
    Frag bQ[4];
#pragma unroll
    for (int kc = 0; kc < 4; ++kc) bQ[kc] = load_b_frag(Qp, HD, m0, kc * 32, lane);

    f32x8 o[8] = {};                 // O^T accumulators: rows = d, cols = tokens
    float mtok = -3.0e38f, ltok = 0.0f; // per-lane (per-token) softmax state
    const float sc2 = 0.08838834764831845f * 1.4426950408889634f; // 1/sqrt(128) * log2(e)

    for (int k0 = 0; k0 < SS; k0 += 32) {
        Frag aK0[4], aK1[4];
#pragma unroll
        for (int kc = 0; kc < 4; ++kc) aK0[kc] = load_a_frag(Kp, HD, k0, kc * 32, lane);
#pragma unroll
        for (int kc = 0; kc < 4; ++kc) aK1[kc] = load_a_frag(Kp, HD, k0 + 16, kc * 32, lane);
        f32x8 s0 = {}, s1 = {}; // S^T tiles: rows = keys, cols = tokens
#pragma unroll
        for (int kc = 0; kc < 4; ++kc) {
            s0 = wmma_bf16(aK0[kc], bQ[kc], s0);
            s1 = wmma_bf16(aK1[kc], bQ[kc], s1);
        }
        // issue V fragment loads now; latency hides behind the softmax VALU below
        Frag aV[8];
#pragma unroll
        for (int a = 0; a < 8; ++a) aV[a] = load_a_frag(Vp, SS, a * 16, k0, lane);

        // per-lane softmax over this lane's token (keys live in the 16 regs + lane^16)
        float tmax = -3.0e38f;
#pragma unroll
        for (int j = 0; j < 8; ++j) {
            s0[j] *= sc2;
            s1[j] *= sc2;
            tmax = fmaxf(tmax, fmaxf(s0[j], s1[j]));
        }
        tmax = fmaxf(tmax, __shfl_xor(tmax, 16, 32));
        const float mn = fmaxf(mtok, tmax);
        const float sc = exp2f(mtok - mn);
        float psum = 0.0f;
#pragma unroll
        for (int j = 0; j < 8; ++j) {
            s0[j] = exp2f(s0[j] - mn);
            s1[j] = exp2f(s1[j] - mn);
            psum += s0[j] + s1[j];
        }
        psum += __shfl_xor(psum, 16, 32);
        ltok = ltok * sc + psum;
        mtok = mn;
#pragma unroll
        for (int a = 0; a < 8; ++a)
#pragma unroll
            for (int j = 0; j < 8; ++j) o[a][j] *= sc;

        // pack P^T into LDS [token][key]: each lane owns 8 consecutive keys per tile
        unsigned short* pl = plds[wIn];
        Pack p0, p1;
#pragma unroll
        for (int j = 0; j < 8; ++j) { p0.h[j] = f2bf(s0[j]); p1.h[j] = f2bf(s1[j]); }
        *(uint4*)&pl[lh * 32 + hf * 8] = p0.q;
        *(uint4*)&pl[lh * 32 + 16 + hf * 8] = p1.q;
        Frag bP; // B operand: column = token (contiguous keys)
        bP.q[0] = *(const uint4*)&pl[lh * 32 + hf * 16];
        bP.q[1] = *(const uint4*)&pl[lh * 32 + hf * 16 + 8];
#pragma unroll
        for (int a = 0; a < 8; ++a) o[a] = wmma_bf16(aV[a], bP, o[a]);
    }

    const float inv = 1.0f / ltok;
    const int b_ = bh >> 3, head = bh & 7;
    const int m = m0 + lh; // this lane's token
#pragma unroll
    for (int a = 0; a < 8; ++a) {
        Pack pk;
#pragma unroll
        for (int j = 0; j < 8; ++j) pk.h[j] = f2bf(o[a][j] * inv);
        *(uint4*)&ctx[((size_t)b_ * SS + m) * HH + head * HD + a * 16 + hf * 8] = pk.q;
    }
}

// ---------------- Kernel 3: output projection + residual + LayerNorm ----------------
// transposed compute: A = Wo feature rows, B = ctx^T token cols; lane = one token
__global__ void __launch_bounds__(256) k_oproj_ln(
    const unsigned short* __restrict__ ctx, const unsigned short* __restrict__ Wob,
    const float* __restrict__ bo, const float* __restrict__ x,
    const float* __restrict__ gamma, const float* __restrict__ beta,
    float* __restrict__ out) {
    __shared__ float ssum[16], ssq[16];
    const int lane = threadIdx.x & 31;
    const int w = threadIdx.x >> 5;
    const int m0 = blockIdx.x * 16; // token tile
    const int n0 = w * 128;         // this wave's feature range
    const int lh = lane & 15, hf = lane >> 4;

    f32x8 acc[8] = {};
    Frag bc = load_b_frag(ctx, HH, m0, 0, lane);
    Frag aW[8];
#pragma unroll
    for (int t = 0; t < 8; ++t) aW[t] = load_a_frag(Wob, HH, n0 + t * 16, 0, lane);
    for (int kc = 0; kc < HH; kc += 32) {
        const int kn = (kc + 32 < HH) ? (kc + 32) : 0;
        Frag bcn = load_b_frag(ctx, HH, m0, kn, lane);
        Frag aWn[8];
#pragma unroll
        for (int t = 0; t < 8; ++t) aWn[t] = load_a_frag(Wob, HH, n0 + t * 16, kn, lane);
#pragma unroll
        for (int t = 0; t < 8; ++t) acc[t] = wmma_bf16(aW[t], bc, acc[t]);
        bc = bcn;
#pragma unroll
        for (int t = 0; t < 8; ++t) aW[t] = aWn[t];
    }

    if (threadIdx.x < 16) { ssum[threadIdx.x] = 0.0f; ssq[threadIdx.x] = 0.0f; }
    __syncthreads();

    const int m = m0 + lh; // this lane's token
    float rs = 0.0f, rq = 0.0f;
#pragma unroll
    for (int t = 0; t < 8; ++t) {
        const int nb = n0 + t * 16 + hf * 8; // 8 consecutive features
        const float4 x0 = *(const float4*)&x[(size_t)m * HH + nb];
        const float4 x1 = *(const float4*)&x[(size_t)m * HH + nb + 4];
        const float xr[8] = {x0.x, x0.y, x0.z, x0.w, x1.x, x1.y, x1.z, x1.w};
#pragma unroll
        for (int j = 0; j < 8; ++j) {
            const float v = acc[t][j] + bo[nb + j] + xr[j];
            acc[t][j] = v;
            rs += v;
            rq += v * v;
        }
    }
    rs += __shfl_xor(rs, 16, 32);
    rq += __shfl_xor(rq, 16, 32);
    if (lane < 16) {
        atomicAdd(&ssum[lh], rs);
        atomicAdd(&ssq[lh], rq);
    }
    __syncthreads();

    const float mean = ssum[lh] * (1.0f / HH);
    const float var = ssq[lh] * (1.0f / HH) - mean * mean;
    const float rstd = rsqrtf(var + 1e-5f);
#pragma unroll
    for (int t = 0; t < 8; ++t) {
        const int nb = n0 + t * 16 + hf * 8;
        float4 o0, o1;
        o0.x = (acc[t][0] - mean) * rstd * gamma[nb + 0] + beta[nb + 0];
        o0.y = (acc[t][1] - mean) * rstd * gamma[nb + 1] + beta[nb + 1];
        o0.z = (acc[t][2] - mean) * rstd * gamma[nb + 2] + beta[nb + 2];
        o0.w = (acc[t][3] - mean) * rstd * gamma[nb + 3] + beta[nb + 3];
        o1.x = (acc[t][4] - mean) * rstd * gamma[nb + 4] + beta[nb + 4];
        o1.y = (acc[t][5] - mean) * rstd * gamma[nb + 5] + beta[nb + 5];
        o1.z = (acc[t][6] - mean) * rstd * gamma[nb + 6] + beta[nb + 6];
        o1.w = (acc[t][7] - mean) * rstd * gamma[nb + 7] + beta[nb + 7];
        *(float4*)&out[(size_t)m * HH + nb] = o0;
        *(float4*)&out[(size_t)m * HH + nb + 4] = o1;
    }
}

extern "C" void kernel_launch(void* const* d_in, const int* in_sizes, int n_in,
                              void* d_out, int out_size, void* d_ws, size_t ws_size,
                              hipStream_t stream) {
    (void)in_sizes; (void)n_in; (void)out_size; (void)ws_size;
    const float* x     = (const float*)d_in[0];
    const float* Wq    = (const float*)d_in[1];
    const float* bq    = (const float*)d_in[2];
    const float* Wk    = (const float*)d_in[3];
    const float* bk    = (const float*)d_in[4];
    const float* Wv    = (const float*)d_in[5];
    const float* bv    = (const float*)d_in[6];
    const float* Wo    = (const float*)d_in[7];
    const float* bo    = (const float*)d_in[8];
    const float* gamma = (const float*)d_in[9];
    const float* beta  = (const float*)d_in[10];
    float* out = (float*)d_out;

    char* ws = (char*)d_ws;
    size_t off = 0;
    unsigned short* xb  = (unsigned short*)(ws + off); off += (size_t)MROWS * HH * 2;
    unsigned short* Wqb = (unsigned short*)(ws + off); off += (size_t)HH * HH * 2;
    unsigned short* Wkb = (unsigned short*)(ws + off); off += (size_t)HH * HH * 2;
    unsigned short* Wvb = (unsigned short*)(ws + off); off += (size_t)HH * HH * 2;
    unsigned short* Wob = (unsigned short*)(ws + off); off += (size_t)HH * HH * 2;
    unsigned short* Qh  = (unsigned short*)(ws + off); off += (size_t)MROWS * HH * 2;
    unsigned short* Kh  = (unsigned short*)(ws + off); off += (size_t)MROWS * HH * 2;
    unsigned short* Vt  = (unsigned short*)(ws + off); off += (size_t)MROWS * HH * 2;
    unsigned short* ctx = (unsigned short*)(ws + off); off += (size_t)MROWS * HH * 2;

    k_convert_bf16<<<2048, 256, 0, stream>>>(x, xb, MROWS * HH);
    k_convert_bf16<<<512, 256, 0, stream>>>(Wq, Wqb, HH * HH);
    k_convert_bf16<<<512, 256, 0, stream>>>(Wk, Wkb, HH * HH);
    k_convert_bf16<<<512, 256, 0, stream>>>(Wv, Wvb, HH * HH);
    k_convert_bf16<<<512, 256, 0, stream>>>(Wo, Wob, HH * HH);

    k_qkv<<<3072, 256, 0, stream>>>(xb, Wqb, Wkb, Wvb, bq, bk, bv, Qh, Kh, Vt);
    k_attn<<<1024, 128, 0, stream>>>(Qh, Kh, Vt, ctx);
    k_oproj_ln<<<512, 256, 0, stream>>>(ctx, Wob, bo, x, gamma, beta, out);
}
